// DotProductAttentionTransformerMD17_51264729645462
// MI455X (gfx1250) — compile-verified, matrix operand order checked by reading, not run on hardware
//
#include <hip/hip_runtime.h>
#include <cstddef>
#include <cstdint>

// ---------------- problem constants ----------------
#define NNODES 10000
#define NEDGES 160000
#define NGRAPH 500
#define DIM_D 480
#define DIM_DOUT 512
#define NB_RBF 128
#define FC_DIM 64
#define NHEAD 4
#define DK_DIM 32
#define DV_DIM 120
#define HD_QK 128   // H*DK
#define HD_V 480    // H*DV
#define NSH_DIM 9
#define NLAYERS 5
#define CUTOFF_R 5.0f
#define INV_SQRT_DK 0.17677669529663687f
#define INV_SQRT_AVGDEG 0.2533537f   // 1/sqrt(15.57930850982666)
#define INV_SQRT_AVGNODES 0.2354931f // 1/sqrt(18.03065905448718)

typedef __attribute__((ext_vector_type(16))) _Float16 v16h;
typedef __attribute__((ext_vector_type(8)))  _Float16 v8h;
typedef __attribute__((ext_vector_type(8)))  float    v8f;
typedef __attribute__((ext_vector_type(4)))  unsigned u32x4;
typedef __attribute__((ext_vector_type(4)))  int      i32x4;
typedef __attribute__((ext_vector_type(8)))  int      i32x8;
typedef int v4i_t __attribute__((vector_size(16)));   // matches async-LDS builtin pointee
#define AS1 __attribute__((address_space(1)))
#define AS3 __attribute__((address_space(3)))

// CDNA5 feature probes (compile-portable: fall back to sync LDS staging)
#if __has_builtin(__builtin_amdgcn_global_load_async_to_lds_b128)
#define HAVE_ASYNC 1
#else
#define HAVE_ASYNC 0
#endif
#if __has_builtin(__builtin_amdgcn_tensor_load_to_lds)
#define HAVE_TDM 1
#else
#define HAVE_TDM 0
#endif

// Input index map (setup_inputs order; params flattened with sorted dict keys)
enum {
  IN_NODE_ATOM = 0, IN_POS, IN_BATCH, IN_ESRC, IN_EDST,
  IN_WH1, IN_WH2, IN_ATOM_EMBED,
  IN_B_WF1, IN_B_WF2, IN_B_WG, IN_B_WK, IN_B_WO, IN_B_WQ,
  IN_B_WR1, IN_B_WR2, IN_B_WSH, IN_B_WV,
  IN_B_LN1B, IN_B_LN1G, IN_B_LN2B, IN_B_LN2G,
  IN_DEG_R1, IN_DEG_R2, IN_DEG_W,
  IN_F_WF1, IN_F_WF2, IN_F_WG, IN_F_WK, IN_F_WO, IN_F_WQ,
  IN_F_WR1, IN_F_WR2, IN_F_WSH, IN_F_WSKIP, IN_F_WV,
  IN_F_LN1B, IN_F_LN1G, IN_F_LN2B, IN_F_LN2G,
  IN_HEAD_LNB, IN_HEAD_LNG
};

// ---------------- generic fills / converts ----------------
__global__ void fill_f32_kernel(float* __restrict__ p, float v, size_t n) {
  size_t i = (size_t)blockIdx.x * blockDim.x + threadIdx.x;
  if (i < n) p[i] = v;
}
__global__ void fill_u32_kernel(unsigned* __restrict__ p, unsigned v, size_t n) {
  size_t i = (size_t)blockIdx.x * blockDim.x + threadIdx.x;
  if (i < n) p[i] = v;
}
__global__ void f32_to_f16_kernel(const float* __restrict__ in, _Float16* __restrict__ out, size_t n) {
  size_t i = (size_t)blockIdx.x * blockDim.x + threadIdx.x;
  if (i < n) out[i] = (_Float16)in[i];
}
// W (K x N, f32, row-major) -> Wt (Np x Kp, f16, row-major) = transposed + zero-padded
__global__ void convert_w_kernel(const float* __restrict__ W, _Float16* __restrict__ Wt,
                                 int K, int N, int Kp, int Np) {
  int idx = blockIdx.x * blockDim.x + threadIdx.x;
  if (idx >= Kp * Np) return;
  int n = idx / Kp, k = idx % Kp;
  Wt[idx] = (k < K && n < N) ? (_Float16)W[(size_t)k * N + n] : (_Float16)0.0f;
}
__global__ void add_f32_kernel(float* __restrict__ X, const float* __restrict__ Y, size_t n) {
  size_t i = (size_t)blockIdx.x * blockDim.x + threadIdx.x;
  if (i < n) X[i] += Y[i];
}

// ---------------- WMMA GEMM: C[M,N] = A[M,K](f16) * Bt[N,K](f16)^T ----------------
// 128 threads = 4 waves in a 2x2 arrangement; 64x64 macro-tile; each wave owns a
// 32x32 sub-tile (2x2 WMMA accumulators). A/B K-slices (64x32 halves each) are
// double-buffered in LDS: A staged by the Tensor Data Mover (TENSORcnt) when
// available, B staged by async global->LDS loads (ASYNCcnt) when available,
// both falling back to plain load + ds_store.
#define GF_SILU 1
#define GF_ACC 2
#define KSTEP 32

__global__ __launch_bounds__(128) void wmma_gemm_kernel(
    const _Float16* __restrict__ A, const _Float16* __restrict__ Bt,
    float* __restrict__ Cf, _Float16* __restrict__ Ch,
    int M, int N, int K, int flags, float alpha)
{
  __shared__ _Float16 lA[2][64 * KSTEP];
  __shared__ _Float16 lB[2][64 * KSTEP];
  const int tid  = threadIdx.x;
  const int wave = tid >> 5, lane = tid & 31;
  const int wm = wave >> 1, wn = wave & 1;
  const int g = lane >> 4, i = lane & 15;
  const int rowBase = blockIdx.x * 64;
  const int colBase = blockIdx.y * 64;

  v8f acc00 = {0.f,0.f,0.f,0.f,0.f,0.f,0.f,0.f};
  v8f acc01 = acc00, acc10 = acc00, acc11 = acc00;

  auto stage = [&](int buf, int k0) {
#if HAVE_TDM
    // ---- A tile via Tensor Data Mover (one descriptor per k-step, wave 0) ----
    if (wave == 0) {
      unsigned ldsOff = (unsigned)(uintptr_t)(AS3 void*)&lA[buf][0];
      unsigned long long gaddr =
          (unsigned long long)(uintptr_t)(A + (size_t)rowBase * K + k0);
      u32x4 G0;
      G0[0] = 1u;                       // count=1, user descriptor
      G0[1] = ldsOff;                   // lds_addr
      G0[2] = (unsigned)gaddr;          // global_addr[31:0]
      G0[3] = (unsigned)((gaddr >> 32) & 0x01FFFFFFu) | (2u << 30); // addr[56:32], type=2
      unsigned td0 = (unsigned)(K - k0);                 // remaining K extent
      int remRows = M - rowBase; if (remRows < 1) remRows = 1;
      unsigned td1 = (unsigned)remRows;                  // remaining row extent (OOB -> 0)
      unsigned long long st0 = (unsigned long long)K;    // dim0 stride (elements)
      i32x8 G1;
      G1[0] = (int)(1u << 16);                           // data_size=1 (2 bytes)
      G1[1] = (int)((td0 & 0xFFFFu) << 16);              // tensor_dim0[15:0]
      G1[2] = (int)((td0 >> 16) | ((td1 & 0xFFFFu) << 16)); // tensor_dim0[31:16] | tensor_dim1[15:0]
      G1[3] = (int)((td1 >> 16) | ((unsigned)KSTEP << 16)); // tensor_dim1[31:16] | tile_dim0=32
      G1[4] = (int)64u;                                  // tile_dim1=64, tile_dim2=0
      G1[5] = (int)(unsigned)st0;                        // tensor_dim0_stride[31:0]
      G1[6] = (int)((unsigned)(st0 >> 32) & 0xFFFFu);    // stride[47:32] | dim1_stride[15:0]=0
      G1[7] = 0;
      i32x4 Z4 = {0, 0, 0, 0};
#if __clang_major__ >= 23
      i32x8 Z8 = {0, 0, 0, 0, 0, 0, 0, 0};
      __builtin_amdgcn_tensor_load_to_lds(G0, G1, Z4, Z4, Z8, 0);
#else
      __builtin_amdgcn_tensor_load_to_lds(G0, G1, Z4, Z4, 0);
#endif
    }
#endif
#pragma unroll
    for (int t = 0; t < 2; ++t) {
      int c = tid + t * 128;            // 0..255 : 64 rows x 4 x 16B chunks
      int row = c >> 2, seg = c & 3;
#if !HAVE_TDM
      int gr = rowBase + row; if (gr >= M) gr = M - 1;
      const _Float16* ga = A + (size_t)gr * K + k0 + seg * 8;
      _Float16* la = &lA[buf][row * KSTEP + seg * 8];
#if HAVE_ASYNC
      __builtin_amdgcn_global_load_async_to_lds_b128(
          (AS1 v4i_t*)ga, (AS3 v4i_t*)la, 0, 0);
#else
      *(v8h*)la = *(const v8h*)ga;
#endif
#endif
      int gc = colBase + row; if (gc >= N) gc = N - 1;
      const _Float16* gb = Bt + (size_t)gc * K + k0 + seg * 8;
      _Float16* lb = &lB[buf][row * KSTEP + seg * 8];
#if HAVE_ASYNC
      __builtin_amdgcn_global_load_async_to_lds_b128(
          (AS1 v4i_t*)gb, (AS3 v4i_t*)lb, 0, 0);
#else
      *(v8h*)lb = *(const v8h*)gb;
#endif
    }
  };

  auto waitStage = [&]() {
#if HAVE_TDM
    if (wave == 0) __builtin_amdgcn_s_wait_tensorcnt(0);
#endif
#if HAVE_ASYNC
#if __has_builtin(__builtin_amdgcn_s_wait_asynccnt)
    __builtin_amdgcn_s_wait_asynccnt(0);
#else
    asm volatile("s_wait_asynccnt 0" ::: "memory");
#endif
#endif
    __syncthreads();
  };

  stage(0, 0);
  waitStage();
  const int nk = K / KSTEP;
  for (int kk = 0; kk < nk; ++kk) {
    const int buf = kk & 1;
    if (kk + 1 < nk) stage(buf ^ 1, (kk + 1) * KSTEP);
    // fragments from LDS (ds_load_b128)
    const int r0 = (wm * 32 + i) * KSTEP;
    const int r1 = (wm * 32 + 16 + i) * KSTEP;
    const int c0 = (wn * 32 + i) * KSTEP;
    const int c1 = (wn * 32 + 16 + i) * KSTEP;
    v8h a0l = *(const v8h*)&lA[buf][r0 + g * 8];
    v8h a0h = *(const v8h*)&lA[buf][r0 + 16 + g * 8];
    v8h a1l = *(const v8h*)&lA[buf][r1 + g * 8];
    v8h a1h = *(const v8h*)&lA[buf][r1 + 16 + g * 8];
    v8h b0l = *(const v8h*)&lB[buf][c0 + g * 8];
    v8h b0h = *(const v8h*)&lB[buf][c0 + 16 + g * 8];
    v8h b1l = *(const v8h*)&lB[buf][c1 + g * 8];
    v8h b1h = *(const v8h*)&lB[buf][c1 + 16 + g * 8];
    v16h a0 = __builtin_shufflevector(a0l, a0h, 0,1,2,3,4,5,6,7,8,9,10,11,12,13,14,15);
    v16h a1 = __builtin_shufflevector(a1l, a1h, 0,1,2,3,4,5,6,7,8,9,10,11,12,13,14,15);
    v16h b0 = __builtin_shufflevector(b0l, b0h, 0,1,2,3,4,5,6,7,8,9,10,11,12,13,14,15);
    v16h b1 = __builtin_shufflevector(b1l, b1h, 0,1,2,3,4,5,6,7,8,9,10,11,12,13,14,15);
    acc00 = __builtin_amdgcn_wmma_f32_16x16x32_f16(false, a0, false, b0, (short)0, acc00, false, false);
    acc01 = __builtin_amdgcn_wmma_f32_16x16x32_f16(false, a0, false, b1, (short)0, acc01, false, false);
    acc10 = __builtin_amdgcn_wmma_f32_16x16x32_f16(false, a1, false, b0, (short)0, acc10, false, false);
    acc11 = __builtin_amdgcn_wmma_f32_16x16x32_f16(false, a1, false, b1, (short)0, acc11, false, false);
    waitStage();
  }

  auto emit = [&](v8f acc, int row0, int col) {
    if (col >= N) return;
#pragma unroll
    for (int r = 0; r < 8; ++r) {
      int row = row0 + r;
      if (row >= M) continue;
      float v = acc[r] * alpha;
      if (flags & GF_SILU) v = v / (1.0f + __expf(-v));
      size_t idx = (size_t)row * N + col;
      if (Ch)                  Ch[idx]  = (_Float16)v;
      else if (flags & GF_ACC) Cf[idx] += v;
      else                     Cf[idx]  = v;
    }
  };
  const int rA = rowBase + wm * 32 + 8 * g;
  const int rB = rowBase + wm * 32 + 16 + 8 * g;
  const int cA = colBase + wn * 32 + i;
  const int cB = colBase + wn * 32 + 16 + i;
  emit(acc00, rA, cA); emit(acc01, rA, cB);
  emit(acc10, rB, cA); emit(acc11, rB, cB);
}

// ---------------- LayerNorm (f32 in -> f16 out), one wave per row ----------------
__global__ __launch_bounds__(64) void layernorm_kernel(
    const float* __restrict__ X, const float* __restrict__ gam, const float* __restrict__ bet,
    _Float16* __restrict__ out, int M, int Dv)
{
  int row = blockIdx.x * 2 + (threadIdx.x >> 5);
  int lane = threadIdx.x & 31;
  if (row >= M) return;
  const float* x = X + (size_t)row * Dv;
  float s = 0.f;
  for (int j = lane; j < Dv; j += 32) s += x[j];
  for (int o = 16; o; o >>= 1) s += __shfl_xor(s, o, 32);
  float mu = s / Dv;
  float v = 0.f;
  for (int j = lane; j < Dv; j += 32) { float t = x[j] - mu; v += t * t; }
  for (int o = 16; o; o >>= 1) v += __shfl_xor(v, o, 32);
  float inv = rsqrtf(v / Dv + 1e-5f);
  _Float16* o16 = out + (size_t)row * Dv;
  for (int j = lane; j < Dv; j += 32)
    o16[j] = (_Float16)((x[j] - mu) * inv * gam[j] + bet[j]);
}

// ---------------- edge geometry: spherical harmonics + RBF ----------------
__global__ void edge_geom_kernel(const float* __restrict__ pos,
    const int* __restrict__ src, const int* __restrict__ dst,
    float* __restrict__ sh, _Float16* __restrict__ rbfh, int E)
{
  int e = blockIdx.x * blockDim.x + threadIdx.x;
  if (e >= E) return;
  int s = src[e], t = dst[e];
  float vx = pos[s*3+0] - pos[t*3+0];
  float vy = pos[s*3+1] - pos[t*3+1];
  float vz = pos[s*3+2] - pos[t*3+2];
  float d = sqrtf(vx*vx + vy*vy + vz*vz + 1e-12f);
  float ux = vx / d, uy = vy / d, uz = vz / d;
  const float s3 = 1.7320508075688772f, s15 = 3.872983346207417f, s5 = 2.23606797749979f;
  float* o = sh + (size_t)e * NSH_DIM;
  o[0] = 1.0f;          o[1] = s3 * ux;       o[2] = s3 * uy;
  o[3] = s3 * uz;       o[4] = s15 * ux * uy; o[5] = s15 * uy * uz;
  o[6] = 0.5f * s5 * (3.0f * uz * uz - 1.0f);
  o[7] = s15 * ux * uz; o[8] = 0.5f * s15 * (ux * ux - uy * uy);
  const float sigma = CUTOFF_R / NB_RBF;
  _Float16* rb = rbfh + (size_t)e * NB_RBF;
  for (int j = 0; j < NB_RBF; ++j) {
    float mu = CUTOFF_R * (float)j / (float)(NB_RBF - 1);
    float z = (d - mu) / sigma;
    rb[j] = (_Float16)__expf(-0.5f * z * z);
  }
}

// ---------------- degree embedding: scatter sh*rw into padded node buffer ----------------
__global__ void srw_scatter_kernel(const float* __restrict__ sh, const float* __restrict__ rw,
    const int* __restrict__ dst, float* __restrict__ srwN, int E)
{
  int idx = blockIdx.x * blockDim.x + threadIdx.x;
  int e = idx / NSH_DIM, j = idx % NSH_DIM;
  if (e >= E) return;
  atomicAdd(&srwN[(size_t)dst[e] * 32 + j], sh[(size_t)e * NSH_DIM + j] * rw[(size_t)e * 16 + j]);
}
__global__ void add_embed_kernel(float* __restrict__ X, const float* __restrict__ emb,
    const int* __restrict__ atom, int M, int Dv)
{
  size_t idx = (size_t)blockIdx.x * blockDim.x + threadIdx.x;
  if (idx >= (size_t)M * Dv) return;
  int n = (int)(idx / Dv), j = (int)(idx % Dv);
  X[idx] += emb[(size_t)atom[n] * Dv + j];
}

// ---------------- attention edge kernels ----------------
__global__ void gsh_kernel(const float* __restrict__ sh, const float* __restrict__ Wsh,
    float* __restrict__ gsh, int E)
{
  int idx = blockIdx.x * blockDim.x + threadIdx.x;
  int e = idx >> 2, h = idx & 3;
  if (e >= E) return;
  const float* s = sh + (size_t)e * NSH_DIM;
  float acc = 0.f;
  for (int k = 0; k < NSH_DIM; ++k) acc += s[k] * Wsh[k * NHEAD + h];
  gsh[idx] = acc;
}

__global__ void logits_kernel(const float* __restrict__ Qf, const float* __restrict__ Kf,
    const float* __restrict__ Gf, const int* __restrict__ src, const int* __restrict__ dst,
    float* __restrict__ logits, unsigned* __restrict__ mmax, int E)
{
  int idx = blockIdx.x * blockDim.x + threadIdx.x;
  int e = idx >> 2, h = idx & 3;
  if (e >= E) return;
  int s = src[e], t = dst[e];
  const float* q = Qf + (size_t)t * HD_QK + h * DK_DIM;
  const float* k = Kf + (size_t)s * HD_QK + h * DK_DIM;
  float acc = 0.f;
  for (int j = 0; j < DK_DIM; ++j) acc += q[j] * k[j];
  float lg = acc * Gf[(size_t)e * 16 + h] * INV_SQRT_DK;
  logits[idx] = lg;
  unsigned b = __float_as_uint(lg);
  unsigned key = (b & 0x80000000u) ? ~b : (b | 0x80000000u);
  atomicMax(&mmax[(size_t)t * NHEAD + h], key);
}

__global__ void zden_kernel(const float* __restrict__ logits, const unsigned* __restrict__ mmax,
    const int* __restrict__ dst, float* __restrict__ zb, float* __restrict__ denom, int E)
{
  int idx = blockIdx.x * blockDim.x + threadIdx.x;
  int e = idx >> 2, h = idx & 3;
  if (e >= E) return;
  int t = dst[e];
  unsigned key = mmax[(size_t)t * NHEAD + h];
  unsigned b = (key & 0x80000000u) ? (key ^ 0x80000000u) : ~key;
  float m = __uint_as_float(b);
  float z = __expf(logits[idx] - m);
  zb[idx] = z;
  atomicAdd(&denom[(size_t)t * NHEAD + h], z);
}

__global__ __launch_bounds__(128) void msg_kernel(const float* __restrict__ zb,
    const float* __restrict__ denom, const float* __restrict__ Vf, const float* __restrict__ gsh,
    const int* __restrict__ src, const int* __restrict__ dst, float* __restrict__ Msg, int E)
{
  int e = blockIdx.x;
  if (e >= E) return;
  int s = src[e], t = dst[e];
  for (int j = threadIdx.x; j < HD_V; j += blockDim.x) {
    int h = j / DV_DIM;
    float alpha = zb[(size_t)e * NHEAD + h] / (denom[(size_t)t * NHEAD + h] + 1e-9f);
    float val = alpha * Vf[(size_t)s * HD_V + j] * gsh[(size_t)e * NHEAD + h];
    atomicAdd(&Msg[(size_t)t * HD_V + j], val);
  }
}

// ---------------- head: e_node dot + per-graph energy ----------------
__global__ __launch_bounds__(64) void enode_kernel(const _Float16* __restrict__ F1h,
    const float* __restrict__ Wh2, const int* __restrict__ batch, float* __restrict__ energy, int M)
{
  int n = blockIdx.x * 2 + (threadIdx.x >> 5);
  int lane = threadIdx.x & 31;
  if (n >= M) return;
  const _Float16* f = F1h + (size_t)n * DIM_DOUT;
  float acc = 0.f;
  for (int j = lane; j < DIM_DOUT; j += 32) acc += (float)f[j] * Wh2[j];
  for (int o = 16; o; o >>= 1) acc += __shfl_xor(acc, o, 32);
  if (lane == 0) atomicAdd(&energy[batch[n]], acc * INV_SQRT_AVGNODES);
}

// ---------------- host-side helpers ----------------
static inline void fillf(hipStream_t st, float* p, float v, size_t n) {
  fill_f32_kernel<<<(unsigned)((n + 255) / 256), 256, 0, st>>>(p, v, n);
}
static inline void fillu(hipStream_t st, unsigned* p, unsigned v, size_t n) {
  fill_u32_kernel<<<(unsigned)((n + 255) / 256), 256, 0, st>>>(p, v, n);
}
static inline void tohalf(hipStream_t st, const float* in, _Float16* out, size_t n) {
  f32_to_f16_kernel<<<(unsigned)((n + 255) / 256), 256, 0, st>>>(in, out, n);
}
static inline void convw(hipStream_t st, const float* W, _Float16* Wt, int K, int N, int Kp, int Np) {
  int n = Kp * Np;
  convert_w_kernel<<<(n + 255) / 256, 256, 0, st>>>(W, Wt, K, N, Kp, Np);
}
static inline void gemm(hipStream_t st, const _Float16* A, const _Float16* Bt,
                        float* Cf, _Float16* Ch, int M, int N, int K, int flags, float alpha) {
  dim3 grid((M + 63) / 64, (N + 63) / 64), block(128);
  wmma_gemm_kernel<<<grid, block, 0, st>>>(A, Bt, Cf, Ch, M, N, K, flags, alpha);
}
static inline void lnorm(hipStream_t st, const float* X, const float* g, const float* b,
                         _Float16* out, int M, int Dv) {
  layernorm_kernel<<<(M + 1) / 2, 64, 0, st>>>(X, g, b, out, M, Dv);
}

extern "C" void kernel_launch(void* const* d_in, const int* in_sizes, int n_in,
                              void* d_out, int out_size, void* d_ws, size_t ws_size,
                              hipStream_t stream)
{
  (void)in_sizes; (void)n_in; (void)out_size; (void)ws_size;
  const int*   node_atom = (const int*)d_in[IN_NODE_ATOM];
  const float* pos       = (const float*)d_in[IN_POS];
  const int*   batch     = (const int*)d_in[IN_BATCH];
  const int*   esrc      = (const int*)d_in[IN_ESRC];
  const int*   edst      = (const int*)d_in[IN_EDST];
  auto F = [&](int i) { return (const float*)d_in[i]; };

  // ---- workspace carve ----
  char* cur = (char*)d_ws;
  auto take = [&](size_t bytes) -> void* {
    void* r = (void*)cur;
    cur += (bytes + 255) & ~(size_t)255;
    return r;
  };
  float*    XA     = (float*)   take((size_t)NNODES * DIM_DOUT * 4);
  float*    XB     = (float*)   take((size_t)NNODES * DIM_DOUT * 4);
  float*    Attn   = (float*)   take((size_t)NNODES * DIM_DOUT * 4); // also Msg scratch
  _Float16* NB1    = (_Float16*)take((size_t)NNODES * DIM_DOUT * 2);
  _Float16* NB2    = (_Float16*)take((size_t)NNODES * DIM_DOUT * 2);
  float*    Qf     = (float*)   take((size_t)NNODES * HD_QK * 4);
  float*    Kf     = (float*)   take((size_t)NNODES * HD_QK * 4);
  float*    Vf     = (float*)   take((size_t)NNODES * HD_V * 4);
  float*    shB    = (float*)   take((size_t)NEDGES * NSH_DIM * 4);
  _Float16* rbfh   = (_Float16*)take((size_t)NEDGES * NB_RBF * 2);
  _Float16* Rh1    = (_Float16*)take((size_t)NEDGES * FC_DIM * 2);
  _Float16* Rh2    = (_Float16*)take((size_t)NEDGES * FC_DIM * 2);
  float*    Gf     = (float*)   take((size_t)NEDGES * 16 * 4);
  float*    gshB   = (float*)   take((size_t)NEDGES * NHEAD * 4);
  float*    logB   = (float*)   take((size_t)NEDGES * NHEAD * 4);
  float*    zB     = (float*)   take((size_t)NEDGES * NHEAD * 4);
  unsigned* mmax   = (unsigned*)take((size_t)NNODES * NHEAD * 4);
  float*    denom  = (float*)   take((size_t)NNODES * NHEAD * 4);
  float*    srwN   = (float*)   take((size_t)NNODES * 32 * 4);
  _Float16* srwNh  = (_Float16*)take((size_t)NNODES * 32 * 2);
  _Float16* wt0    = (_Float16*)take((size_t)512 * 512 * 2);

  float* out_f = (float*)d_out;
  fillf(stream, out_f, 0.0f, (size_t)NGRAPH + (size_t)NNODES * 3); // energy + forces(zeroed)

  // ---- edge geometry: sh (f32) and rbf (f16) ----
  edge_geom_kernel<<<(NEDGES + 255) / 256, 256, 0, stream>>>(pos, esrc, edst, shB, rbfh, NEDGES);

  // ---- degree embedding: rw = silu(rbf@deg_r1)@deg_r2, deg = seg_sum(sh*rw)@deg_w ----
  convw(stream, F(IN_DEG_R1), wt0, NB_RBF, FC_DIM, NB_RBF, FC_DIM);
  gemm(stream, rbfh, wt0, nullptr, Rh1, NEDGES, FC_DIM, NB_RBF, GF_SILU, 1.0f);
  convw(stream, F(IN_DEG_R2), wt0, FC_DIM, NSH_DIM, FC_DIM, 16); // pad N 9->16
  gemm(stream, Rh1, wt0, Gf, nullptr, NEDGES, 16, FC_DIM, 0, 1.0f);
  fillf(stream, srwN, 0.0f, (size_t)NNODES * 32);
  srw_scatter_kernel<<<(NEDGES * NSH_DIM + 255) / 256, 256, 0, stream>>>(shB, Gf, edst, srwN, NEDGES);
  tohalf(stream, srwN, srwNh, (size_t)NNODES * 32);
  convw(stream, F(IN_DEG_W), wt0, NSH_DIM, DIM_D, 32, DIM_D); // pad K 9->32
  gemm(stream, srwNh, wt0, XA, nullptr, NNODES, DIM_D, 32, 0, INV_SQRT_AVGDEG);
  add_embed_kernel<<<(unsigned)(((size_t)NNODES * DIM_D + 255) / 256), 256, 0, stream>>>(
      XA, F(IN_ATOM_EMBED), node_atom, NNODES, DIM_D);

  // ---- attention block runner ----
  auto runBlock = [&](const float* Xin, float* Xout, int din, int dout,
                      const float* Wq, const float* Wk, const float* Wv, const float* Wo,
                      const float* Wr1, const float* Wr2, const float* Wg, const float* Wsh,
                      const float* ln1g, const float* ln1b, const float* ln2g, const float* ln2b,
                      const float* Wf1, const float* Wf2, const float* Wskip)
  {
    // h = LN(x)
    lnorm(stream, Xin, ln1g, ln1b, NB1, NNODES, din);
    // node-level projections (gathered per edge later — identical math to ref)
    convw(stream, Wq, wt0, din, HD_QK, din, HD_QK);
    gemm(stream, NB1, wt0, Qf, nullptr, NNODES, HD_QK, din, 0, 1.0f);
    convw(stream, Wk, wt0, din, HD_QK, din, HD_QK);
    gemm(stream, NB1, wt0, Kf, nullptr, NNODES, HD_QK, din, 0, 1.0f);
    convw(stream, Wv, wt0, din, HD_V, din, HD_V);
    gemm(stream, NB1, wt0, Vf, nullptr, NNODES, HD_V, din, 0, 1.0f);
    // edge radial gate: r = silu(silu(rbf@Wr1)@Wr2); g = r@Wg (N padded 4->16)
    convw(stream, Wr1, wt0, NB_RBF, FC_DIM, NB_RBF, FC_DIM);
    gemm(stream, rbfh, wt0, nullptr, Rh1, NEDGES, FC_DIM, NB_RBF, GF_SILU, 1.0f);
    convw(stream, Wr2, wt0, FC_DIM, FC_DIM, FC_DIM, FC_DIM);
    gemm(stream, Rh1, wt0, nullptr, Rh2, NEDGES, FC_DIM, FC_DIM, GF_SILU, 1.0f);
    convw(stream, Wg, wt0, FC_DIM, NHEAD, FC_DIM, 16);
    gemm(stream, Rh2, wt0, Gf, nullptr, NEDGES, 16, FC_DIM, 0, 1.0f);
    // sh gate per head
    gsh_kernel<<<(NEDGES * NHEAD + 255) / 256, 256, 0, stream>>>(shB, Wsh, gshB, NEDGES);
    // segment softmax
    fillu(stream, mmax, 0x007FFFFFu, (size_t)NNODES * NHEAD); // mapped(-inf)
    fillf(stream, denom, 0.0f, (size_t)NNODES * NHEAD);
    logits_kernel<<<(NEDGES * NHEAD + 255) / 256, 256, 0, stream>>>(
        Qf, Kf, Gf, esrc, edst, logB, mmax, NEDGES);
    zden_kernel<<<(NEDGES * NHEAD + 255) / 256, 256, 0, stream>>>(
        logB, mmax, edst, zB, denom, NEDGES);
    // message aggregation (Attn buffer reused as Msg f32)
    fillf(stream, Attn, 0.0f, (size_t)NNODES * HD_V);
    msg_kernel<<<NEDGES, 128, 0, stream>>>(zB, denom, Vf, gshB, esrc, edst, Attn, NEDGES);
    tohalf(stream, Attn, NB1, (size_t)NNODES * HD_V);
    // attn = msg @ Wo ; + skip
    convw(stream, Wo, wt0, HD_V, dout, HD_V, dout);
    if (Wskip) {
      gemm(stream, NB1, wt0, Xout, nullptr, NNODES, dout, HD_V, 0, 1.0f);
      tohalf(stream, Xin, NB2, (size_t)NNODES * din);
      convw(stream, Wskip, wt0, din, dout, din, dout);
      gemm(stream, NB2, wt0, Xout, nullptr, NNODES, dout, din, GF_ACC, 1.0f);
    } else {
      gemm(stream, NB1, wt0, Attn, nullptr, NNODES, dout, HD_V, 0, 1.0f);
      add_f32_kernel<<<(unsigned)(((size_t)NNODES * dout + 255) / 256), 256, 0, stream>>>(
          Xout, Attn, (size_t)NNODES * dout);
    }
    // FFN: x += silu(LN(x)@Wf1)@Wf2
    lnorm(stream, Xout, ln2g, ln2b, NB1, NNODES, dout);
    convw(stream, Wf1, wt0, dout, dout, dout, dout);
    gemm(stream, NB1, wt0, nullptr, NB2, NNODES, dout, dout, GF_SILU, 1.0f);
    convw(stream, Wf2, wt0, dout, dout, dout, dout);
    gemm(stream, NB2, wt0, Xout, nullptr, NNODES, dout, dout, GF_ACC, 1.0f);
  };

  // ---- 5 regular blocks (stacked weights, leading dim = layer) ----
  for (int l = 0; l < NLAYERS; ++l) {
    size_t oQK = (size_t)l * DIM_D * HD_QK, oVV = (size_t)l * DIM_D * HD_V;
    runBlock(XA, XA, DIM_D, DIM_D,
             F(IN_B_WQ) + oQK, F(IN_B_WK) + oQK, F(IN_B_WV) + oVV, F(IN_B_WO) + oVV,
             F(IN_B_WR1) + (size_t)l * NB_RBF * FC_DIM,
             F(IN_B_WR2) + (size_t)l * FC_DIM * FC_DIM,
             F(IN_B_WG)  + (size_t)l * FC_DIM * NHEAD,
             F(IN_B_WSH) + (size_t)l * NSH_DIM * NHEAD,
             F(IN_B_LN1G) + (size_t)l * DIM_D, F(IN_B_LN1B) + (size_t)l * DIM_D,
             F(IN_B_LN2G) + (size_t)l * DIM_D, F(IN_B_LN2B) + (size_t)l * DIM_D,
             F(IN_B_WF1) + (size_t)l * DIM_D * DIM_D,
             F(IN_B_WF2) + (size_t)l * DIM_D * DIM_D,
             nullptr);
  }
  // ---- final block (480 -> 512, with Wskip) ----
  runBlock(XA, XB, DIM_D, DIM_DOUT,
           F(IN_F_WQ), F(IN_F_WK), F(IN_F_WV), F(IN_F_WO),
           F(IN_F_WR1), F(IN_F_WR2), F(IN_F_WG), F(IN_F_WSH),
           F(IN_F_LN1G), F(IN_F_LN1B), F(IN_F_LN2G), F(IN_F_LN2B),
           F(IN_F_WF1), F(IN_F_WF2), F(IN_F_WSKIP));

  // ---- head: energy = seg_sum(silu(LN(x)@Wh1)@Wh2) / sqrt(avg_nodes) ----
  lnorm(stream, XB, F(IN_HEAD_LNG), F(IN_HEAD_LNB), NB1, NNODES, DIM_DOUT);
  convw(stream, F(IN_WH1), wt0, DIM_DOUT, DIM_DOUT, DIM_DOUT, DIM_DOUT);
  gemm(stream, NB1, wt0, nullptr, NB2, NNODES, DIM_DOUT, DIM_DOUT, GF_SILU, 1.0f);
  enode_kernel<<<(NNODES + 1) / 2, 64, 0, stream>>>(NB2, F(IN_WH2), batch, out_f, NNODES);
  // forces region (out_f + NGRAPH .. +3*NNODES) already zero-filled above.
}
